// HybridAttentionLayer_24575802868094
// MI455X (gfx1250) — matrix operand order repaired and checked by reading, not compiled, not run on hardware
//
#include <hip/hip_runtime.h>
#include <cstdint>

// ---------------------------------------------------------------------------
// HybridAttentionLayer for MI455X (gfx1250, wave32, WMMA)
//   B=4, C=256, C8=32, N=64*64=4096
//   K0 convert : x (f32) -> xT [B][N][C] bf16 ; pack Wq|Wk|Wv -> Wbf [320][256]
//   K1 projW   : WMMA GEMM (software-pipelined B-fragments)
//                -> qT [B][N][32], kT [B][N][32], v [B][C][N]   (bf16)
//   K2 flash   : two-pass flash attention; S^T = K Q^T so softmax state is
//                per-lane and P lands directly in A-fragment layout.
//                K/V fragment loads software-pipelined across WMMAs.
//   K3 eca     : per-(b,c) mean -> y = sigmoid(w_center * mean)
//   K4 scale   : d_out *= y[b,c]
// ---------------------------------------------------------------------------

#define B_  4
#define C_  256
#define D_  32
#define N_  4096
#define O_  320            // 32 q + 32 k + 256 v output rows
#define XTOT (B_ * C_ * N_)
#define NT_ (N_ / 32)      // 128 key tiles

typedef __attribute__((ext_vector_type(16))) __bf16 bf16x16;
typedef __attribute__((ext_vector_type(8)))  float  f32x8;

struct Two128 { uint4 a, b; };

static __device__ __forceinline__ bf16x16 ld_frag(const uint16_t* p0, const uint16_t* p1) {
  union { Two128 u; bf16x16 v; } f;
  f.u.a = *(const uint4*)p0;   // 8 bf16 (16B)
  f.u.b = *(const uint4*)p1;   // 8 bf16 (16B)
  return f.v;
}

static __device__ __forceinline__ uint16_t f2bf(float x) {
  union { float f; uint32_t u; } cv; cv.f = x;
  uint32_t r = cv.u + 0x7fffu + ((cv.u >> 16) & 1u);   // RNE
  return (uint16_t)(r >> 16);
}

// ---------------- K0: convert/transpose x, pack weights ----------------
__global__ __launch_bounds__(256)
void convert_kernel(const float* __restrict__ x,
                    const float* __restrict__ Wq, const float* __restrict__ Wk,
                    const float* __restrict__ Wv,
                    uint16_t* __restrict__ xT, uint16_t* __restrict__ Wbf) {
  int tid = blockIdx.x * blockDim.x + threadIdx.x;
  if (tid < XTOT) {
    int n = tid & (N_ - 1);
    int c = (tid >> 12) & (C_ - 1);
    int b = tid >> 20;
    xT[((size_t)(b * N_ + n)) * C_ + c] = f2bf(x[tid]);
  } else if (tid < XTOT + O_ * C_) {
    int t = tid - XTOT;
    int o = t >> 8;
    int c = t & (C_ - 1);
    float w;
    if (o < 32)       w = Wq[(size_t)o * C_ + c];
    else if (o < 64)  w = Wk[(size_t)(o - 32) * C_ + c];
    else              w = Wv[(size_t)(o - 64) * C_ + c];
    Wbf[t] = f2bf(w);
  }
}

// ---------------- K1: WMMA projection GEMM (pipelined) ----------------
__global__ __launch_bounds__(128)
void proj_kernel(const uint16_t* __restrict__ Wbf,
                 const uint16_t* __restrict__ xT,
                 const float* __restrict__ bq, const float* __restrict__ bk,
                 const float* __restrict__ bv,
                 uint16_t* __restrict__ qT, uint16_t* __restrict__ kT,
                 uint16_t* __restrict__ v) {
  const int lane  = threadIdx.x & 31;
  // wave-uniform ids -> force SGPR so branches/addresses are scalar
  const int strip = __builtin_amdgcn_readfirstlane(blockIdx.x * 4 + (threadIdx.x >> 5));
  const int nGrp  = strip & 63;
  const int oT    = (strip >> 6) % (O_ / 16);
  const int b     = strip / (64 * (O_ / 16));
  const int obase = oT * 16;
  const int hi    = lane >> 4;
  const int nl    = lane & 15;
  const int abase = hi * 8;
  const int kb    = hi * 16;

  bf16x16 a[8];
#pragma unroll
  for (int kk = 0; kk < 8; ++kk) {
    const uint16_t* wr = Wbf + (size_t)(obase + nl) * C_ + kk * 32;
    a[kk] = ld_frag(wr + abase, wr + abase + 16);
  }

  float bias[8];
#pragma unroll
  for (int r = 0; r < 8; ++r) {
    int o = obase + r + hi * 8;
    bias[r] = (obase < 32) ? bq[o] : (obase < 64) ? bk[o - 32] : bv[o - 64];
  }

  const uint16_t* xTb = xT + (size_t)b * N_ * C_;
  const int nbase0 = nGrp * 64;

  auto loadB = [&](int step) -> bf16x16 {
    int nt = step >> 3, kk = step & 7;
    const uint16_t* xr = xTb + (size_t)(nbase0 + nt * 16 + nl) * C_ + kk * 32 + abase;
    return ld_frag(xr, xr + 16);
  };

  // 32-step flat (nt,kk) pipeline, triple-buffered B fragments
  bf16x16 b0 = loadB(0);
  bf16x16 b1 = loadB(1);
  f32x8 acc = (f32x8){};
#pragma unroll
  for (int step = 0; step < 32; ++step) {
    const int nt = step >> 3, kk = step & 7;
    bf16x16 bn = loadB(step + 2 < 32 ? step + 2 : 0);
    acc = __builtin_amdgcn_wmma_f32_16x16x32_bf16(false, a[kk], false, b0,
                                                  (short)0, acc, false, false);
    b0 = b1; b1 = bn;
    if (kk == 7) {
      const int nbase = nbase0 + nt * 16;
      if (obase < 64) {
        // q/k: 8 contiguous halves per lane -> one b128 store
        union { uint16_t h[8]; uint4 u; } pk;
#pragma unroll
        for (int r = 0; r < 8; ++r) pk.h[r] = f2bf(acc[r] + bias[r]);
        uint16_t* base = (obase < 32) ? qT : kT;
        int oo = (obase & 31) + hi * 8;
        *(uint4*)(base + ((size_t)(b * N_ + nbase + nl)) * D_ + oo) = pk.u;
      } else {
#pragma unroll
        for (int r = 0; r < 8; ++r) {
          int c = obase - 64 + r + hi * 8;
          v[((size_t)(b * C_ + c)) * N_ + nbase + nl] = f2bf(acc[r] + bias[r]);
        }
      }
      acc = (f32x8){};
    }
  }
}

// ---------------- K2: two-pass flash attention + residual ----------------
__global__ __launch_bounds__(128)
void flash_kernel(const uint16_t* __restrict__ qT,
                  const uint16_t* __restrict__ kT,
                  const uint16_t* __restrict__ vv,
                  const float* __restrict__ x,
                  const float* __restrict__ gamma,
                  float* __restrict__ out) {
  const int lane  = threadIdx.x & 31;
  const int tile  = __builtin_amdgcn_readfirstlane(blockIdx.x * 4 + (threadIdx.x >> 5));
  const int b     = tile >> 8;
  const int ibase = (tile & 255) << 4;
  const int hi    = lane >> 4;
  const int nl    = lane & 15;
  const int abase = hi * 8;
  const int kb    = hi * 16;

  const uint16_t* qrow = qT + ((size_t)(b * N_ + ibase + nl)) * D_ + abase;
  const bf16x16 qB = ld_frag(qrow, qrow + 16);

  const uint16_t* kTb = kT + (size_t)b * N_ * D_;
  const uint16_t* vb  = vv + (size_t)b * C_ * N_;
  const float g = gamma[0];
  const f32x8 z = (f32x8){};

  auto loadK = [&](int jt, bf16x16& a0, bf16x16& a1) {
    const uint16_t* k0 = kTb + (size_t)(jt * 32 + nl) * D_;
    const uint16_t* k1 = kTb + (size_t)(jt * 32 + 16 + nl) * D_;
    a0 = ld_frag(k0 + abase, k0 + abase + 16);
    a1 = ld_frag(k1 + abase, k1 + abase + 16);
  };

  // ---------- pass 1: row max + sum(exp); K frags double-buffered ---------
  bf16x16 kA0, kA1, kA0n, kA1n;
  loadK(0, kA0, kA1);
  float rm = -3.0e38f, rs = 0.0f;
  for (int jt = 0; jt < NT_; ++jt) {
    loadK(jt + 1 < NT_ ? jt + 1 : jt, kA0n, kA1n);   // in flight during WMMA+VALU
    f32x8 st0 = __builtin_amdgcn_wmma_f32_16x16x32_bf16(false, kA0, false, qB,
                                                        (short)0, z, false, false);
    f32x8 st1 = __builtin_amdgcn_wmma_f32_16x16x32_bf16(false, kA1, false, qB,
                                                        (short)0, z, false, false);
    float m = fmaxf(st0[0], st1[0]);
#pragma unroll
    for (int r = 1; r < 8; ++r) m = fmaxf(m, fmaxf(st0[r], st1[r]));
    m = fmaxf(m, __shfl_xor(m, 16, 32));             // combine partner half
    float mnew = fmaxf(rm, m);
    float ps = 0.0f;
#pragma unroll
    for (int r = 0; r < 8; ++r)
      ps += __expf(st0[r] - mnew) + __expf(st1[r] - mnew);
    ps += __shfl_xor(ps, 16, 32);
    rs = rs * __expf(rm - mnew) + ps;
    rm = mnew;
    kA0 = kA0n; kA1 = kA1n;
  }
  const float invl = 1.0f / rs;

  // ---------- pass 2: O = (exp(S-m)/l) V ; V frags triple-buffered --------
  f32x8 acc[16];
#pragma unroll
  for (int t = 0; t < 16; ++t) acc[t] = (f32x8){};

  auto loadV = [&](int jbase, int t) -> bf16x16 {
    const uint16_t* vr = vb + (size_t)(t * 16 + nl) * N_ + jbase + abase;
    return ld_frag(vr, vr + 16);
  };

  loadK(0, kA0, kA1);
  for (int jt = 0; jt < NT_; ++jt) {
    const int jbase = jt * 32;
    loadK(jt + 1 < NT_ ? jt + 1 : jt, kA0n, kA1n);   // fly during S+PV chain
    f32x8 st0 = __builtin_amdgcn_wmma_f32_16x16x32_bf16(false, kA0, false, qB,
                                                        (short)0, z, false, false);
    f32x8 st1 = __builtin_amdgcn_wmma_f32_16x16x32_bf16(false, kA1, false, qB,
                                                        (short)0, z, false, false);
    // S^T D-layout == P A-layout: element e<8 -> st0[e], e>=8 -> st1[e-8]
    bf16x16 pA;
#pragma unroll
    for (int e = 0; e < 8; ++e) {
      pA[e]     = (__bf16)(__expf(st0[e] - rm) * invl);
      pA[e + 8] = (__bf16)(__expf(st1[e] - rm) * invl);
    }
    bf16x16 vB0 = loadV(jbase, 0);
    bf16x16 vB1 = loadV(jbase, 1);
#pragma unroll
    for (int t = 0; t < 16; ++t) {
      bf16x16 vBn = loadV(jbase, t + 2 < 16 ? t + 2 : 0);
      acc[t] = __builtin_amdgcn_wmma_f32_16x16x32_bf16(false, pA, false, vB0,
                                                       (short)0, acc[t],
                                                       false, false);
      vB0 = vB1; vB1 = vBn;
    }
    kA0 = kA0n; kA1 = kA1n;
  }

  // epilogue: residual + gamma (acc already normalized)
#pragma unroll
  for (int r = 0; r < 8; ++r) {
    int n = ibase + r + hi * 8;
#pragma unroll
    for (int t = 0; t < 16; ++t) {
      int c = t * 16 + nl;
      size_t idx = ((size_t)(b * C_ + c) << 12) + n;
      out[idx] = fmaf(g, acc[t][r], x[idx]);
    }
  }
}

// ---------------- K3: ECA gate (mean -> sigmoid) ----------------
__global__ __launch_bounds__(256)
void eca_reduce_kernel(const float* __restrict__ out,
                       const float* __restrict__ w_eca,
                       float* __restrict__ y) {
  __shared__ float red[256];
  int bc = blockIdx.x;
  int c  = bc & (C_ - 1);
  const float* row = out + (size_t)bc * N_;
  float s = 0.0f;
  for (int i = threadIdx.x; i < N_; i += 256) s += row[i];
  red[threadIdx.x] = s;
  __syncthreads();
  for (int off = 128; off > 0; off >>= 1) {
    if (threadIdx.x < off) red[threadIdx.x] += red[threadIdx.x + off];
    __syncthreads();
  }
  if (threadIdx.x == 0) {
    float mean = red[0] * (1.0f / (float)N_);
    float t = w_eca[c * 3 + 1] * mean;     // center tap of k=3 dw-conv
    y[bc] = 1.0f / (1.0f + __expf(-t));
  }
}

// ---------------- K4: apply gate ----------------
__global__ __launch_bounds__(256)
void eca_scale_kernel(float* __restrict__ out, const float* __restrict__ y) {
  size_t idx = (size_t)blockIdx.x * blockDim.x + threadIdx.x;
  if (idx >= (size_t)B_ * C_ * N_) return;
  out[idx] *= y[idx >> 12];
}

// ---------------------------------------------------------------------------
extern "C" void kernel_launch(void* const* d_in, const int* in_sizes, int n_in,
                              void* d_out, int out_size, void* d_ws, size_t ws_size,
                              hipStream_t stream) {
  const float* x     = (const float*)d_in[0];
  const float* Wq    = (const float*)d_in[1];
  const float* bq    = (const float*)d_in[2];
  const float* Wk    = (const float*)d_in[3];
  const float* bk    = (const float*)d_in[4];
  const float* Wv    = (const float*)d_in[5];
  const float* bv    = (const float*)d_in[6];
  const float* gamma = (const float*)d_in[7];
  const float* weca  = (const float*)d_in[8];
  float* out = (float*)d_out;

  uint16_t* xT  = (uint16_t*)d_ws;
  uint16_t* Wbf = xT  + (size_t)XTOT;
  uint16_t* qT  = Wbf + (size_t)O_ * C_;
  uint16_t* kT  = qT  + (size_t)B_ * N_ * D_;
  uint16_t* vv  = kT  + (size_t)B_ * N_ * D_;
  float*    y   = (float*)(vv + (size_t)B_ * C_ * N_);

  {
    int total = XTOT + O_ * C_;
    convert_kernel<<<(total + 255) / 256, 256, 0, stream>>>(x, Wq, Wk, Wv, xT, Wbf);
  }
  {
    proj_kernel<<<1280, 128, 0, stream>>>(Wbf, xT, bq, bk, bv, qT, kT, vv);
  }
  {
    flash_kernel<<<256, 128, 0, stream>>>(qT, kT, vv, x, gamma, out);
  }
  {
    eca_reduce_kernel<<<B_ * C_, 256, 0, stream>>>(out, weca, y);
  }
  {
    int total = B_ * C_ * N_;
    eca_scale_kernel<<<(total + 255) / 256, 256, 0, stream>>>(out, y);
  }
}
